// GCN_49100066128549
// MI455X (gfx1250) — matrix operand order repaired and checked by reading, not compiled, not run on hardware
//
#include <hip/hip_runtime.h>
#include <hip/hip_bf16.h>

#define N_NODES 50000
#define N_EDGES 1200000
#define IN_DIM  256
#define HID_DIM 64
#define TILES_M (N_NODES / 16)   // 3125, exact
#define BN_EPS  1e-5f

typedef __attribute__((ext_vector_type(16))) _Float16 v16h;
typedef __attribute__((ext_vector_type(8)))  float    v8f;
typedef int v4i_vec __attribute__((vector_size(16)));

#ifndef __has_builtin
#define __has_builtin(x) 0
#endif

#if __has_builtin(__builtin_amdgcn_global_load_async_to_lds_b128) && \
    __has_builtin(__builtin_amdgcn_s_wait_asynccnt)
#define USE_ASYNC_LDS 1
#else
#define USE_ASYNC_LDS 0
#endif

// ---------------------------------------------------------------------------
// Zero the buffers we accumulate into (deg + BN stats).
// ---------------------------------------------------------------------------
__global__ void zero_k(float* __restrict__ deg, float* __restrict__ stats) {
    int i = blockIdx.x * blockDim.x + threadIdx.x;
    if (i < N_NODES) deg[i] = 0.0f;
    if (i < 256)     stats[i] = 0.0f;
}

// ---------------------------------------------------------------------------
// GEMM1: h1[50000,64] = x[50000,256] @ W1[256,64]   (WMMA f16 -> f32 acc)
// One wave = one 16x64 output tile.
// W1 is staged to LDS with gfx1250 async-to-LDS copies (ASYNCcnt), then
// swizzled LDS->LDS into a fragment-major f16 layout so each B operand is
// two contiguous ds_load_b128s.
// ---------------------------------------------------------------------------
__global__ __launch_bounds__(256) void gemm1_wmma_k(
    const float* __restrict__ x, const float* __restrict__ W1,
    float* __restrict__ h1) {
    // fragment-major W1 in f16: [kblock(8)][ntile(4)][lane(32)][16 halfs] = 32 KB
    __shared__ _Float16 ldsW[IN_DIM * HID_DIM];
    const int tid = threadIdx.x;

#if USE_ASYNC_LDS
    __shared__ float stageW[IN_DIM * HID_DIM];   // 64 KB raw f32 copy
    // 16384 floats = 4096 x 16B chunks; per-lane async copies, ASYNCcnt-tracked.
    for (int c = tid; c < (IN_DIM * HID_DIM) / 4; c += 256) {
        __builtin_amdgcn_global_load_async_to_lds_b128(
            (__attribute__((address_space(1))) v4i_vec*)(W1 + c * 4),
            (__attribute__((address_space(3))) v4i_vec*)(&stageW[c * 4]),
            0, 0);
    }
    __builtin_amdgcn_s_wait_asynccnt(0);  // this wave's copies landed in LDS
    __syncthreads();                      // cross-wave visibility
    const float* __restrict__ Wsrc = stageW;
#else
    const float* __restrict__ Wsrc = W1;
#endif

    // f32 -> f16 fragment-major swizzle
    for (int idx = tid; idx < IN_DIM * HID_DIM; idx += 256) {
        int K = idx >> 6;          // 0..255 (row of W1)
        int c = idx & 63;          // 0..63  (col of W1)
        _Float16 val = (_Float16)Wsrc[idx];
        int kb = K >> 5;           // k-block 0..7
        int j  = (K & 7) + ((K >> 4) & 1) * 8;  // half index inside fragment
        int kh = (K >> 3) & 1;     // lane half: K%16 in [0,8) -> lanes 0-15
        int t  = c >> 4;           // n-tile 0..3
        int n  = c & 15;
        int L  = kh * 16 + n;      // lane that owns this element
        ldsW[((kb * 4 + t) * 32 + L) * 16 + j] = val;
    }
    __syncthreads();

    const int lane = tid & 31;
    const int tile = blockIdx.x * 8 + (tid >> 5);
    if (tile >= TILES_M) return;   // wave-uniform guard: EXEC stays all-ones

    const int m     = lane & 15;
    const int kbase = (lane < 16) ? 0 : 8;     // A-row half selector
    const float* xrow = x + (size_t)(tile * 16 + m) * IN_DIM;

    v8f acc0 = {}, acc1 = {}, acc2 = {}, acc3 = {};

    #pragma unroll
    for (int kb = 0; kb < 8; ++kb) {
        const int k0 = kb * 32;
        // A fragment: lane<16 holds K {k0..k0+7, k0+16..k0+23}; lane>=16 -> +8
        const float4 f0 = *(const float4*)(xrow + k0 + kbase);
        const float4 f1 = *(const float4*)(xrow + k0 + kbase + 4);
        const float4 f2 = *(const float4*)(xrow + k0 + 16 + kbase);
        const float4 f3 = *(const float4*)(xrow + k0 + 16 + kbase + 4);
        v16h a;
        a[0]=(_Float16)f0.x;  a[1]=(_Float16)f0.y;  a[2]=(_Float16)f0.z;  a[3]=(_Float16)f0.w;
        a[4]=(_Float16)f1.x;  a[5]=(_Float16)f1.y;  a[6]=(_Float16)f1.z;  a[7]=(_Float16)f1.w;
        a[8]=(_Float16)f2.x;  a[9]=(_Float16)f2.y;  a[10]=(_Float16)f2.z; a[11]=(_Float16)f2.w;
        a[12]=(_Float16)f3.x; a[13]=(_Float16)f3.y; a[14]=(_Float16)f3.z; a[15]=(_Float16)f3.w;

        const _Float16* wbase = &ldsW[(kb * 4) * 32 * 16 + lane * 16];
        v16h b0 = *(const v16h*)(wbase + 0 * 32 * 16);
        v16h b1 = *(const v16h*)(wbase + 1 * 32 * 16);
        v16h b2 = *(const v16h*)(wbase + 2 * 32 * 16);
        v16h b3 = *(const v16h*)(wbase + 3 * 32 * 16);

        acc0 = __builtin_amdgcn_wmma_f32_16x16x32_f16(false, a, false, b0, (short)0, acc0, false, false);
        acc1 = __builtin_amdgcn_wmma_f32_16x16x32_f16(false, a, false, b1, (short)0, acc1, false, false);
        acc2 = __builtin_amdgcn_wmma_f32_16x16x32_f16(false, a, false, b2, (short)0, acc2, false, false);
        acc3 = __builtin_amdgcn_wmma_f32_16x16x32_f16(false, a, false, b3, (short)0, acc3, false, false);
    }

    // D layout: lane holds column n = lane%16, rows m = r + 8*(lane>=16)
    const int n   = lane & 15;
    const int mhi = (lane >> 4) * 8;
    float* ob = h1 + (size_t)(tile * 16 + mhi) * HID_DIM + n;
    #pragma unroll
    for (int r = 0; r < 8; ++r) {
        float* row = ob + (size_t)r * HID_DIM;
        row[0]  = acc0[r];
        row[16] = acc1[r];
        row[32] = acc2[r];
        row[48] = acc3[r];
    }
}

// ---------------------------------------------------------------------------
// In-degree (dst side) over edges; self-loop handled as +1 in dinv.
// ---------------------------------------------------------------------------
__global__ void deg_k(const int* __restrict__ dst, float* __restrict__ deg) {
    int e = blockIdx.x * blockDim.x + threadIdx.x;
    if (e < N_EDGES) atomicAdd(&deg[dst[e]], 1.0f);
}

__global__ void dinv_k(const float* __restrict__ deg, float* __restrict__ dinv) {
    int i = blockIdx.x * blockDim.x + threadIdx.x;
    if (i < N_NODES) dinv[i] = rsqrtf(deg[i] + 1.0f);
}

// agg1 = h1 * dinv^2 (self-loop message) + b1   (init before edge scatter)
__global__ void selfinit_k(const float* __restrict__ h1, const float* __restrict__ dinv,
                           const float* __restrict__ b1, float* __restrict__ agg1) {
    int idx = blockIdx.x * blockDim.x + threadIdx.x;
    if (idx < N_NODES * HID_DIM) {
        int i = idx >> 6, d = idx & 63;
        float w = dinv[i];
        agg1[idx] = h1[idx] * w * w + b1[d];
    }
}

// Edge scatter, 64-dim messages: 16 threads per edge, float4 gather + 4 atomics.
// One lane in 16 prefetches the streaming edge arrays a few KB ahead
// (global_prefetch_b8); the h1 gather is random but L2-resident.
__global__ void scatter1_k(const int* __restrict__ src, const int* __restrict__ dst,
                           const float* __restrict__ dinv,
                           const float* __restrict__ h1, float* __restrict__ agg1) {
    int tid = blockIdx.x * blockDim.x + threadIdx.x;
    int e = tid >> 4;
    if (e >= N_EDGES) return;
    int g = tid & 15;
    if (g == 0 && e + 4096 < N_EDGES) {
        __builtin_prefetch(src + e + 4096, 0, 0);
        __builtin_prefetch(dst + e + 4096, 0, 0);
    }
    int s = src[e], d = dst[e];
    float w = dinv[s] * dinv[d];
    const float4 v = *(const float4*)(h1 + (size_t)s * HID_DIM + g * 4);
    float* o = agg1 + (size_t)d * HID_DIM + g * 4;
    atomicAdd(o + 0, v.x * w);
    atomicAdd(o + 1, v.y * w);
    atomicAdd(o + 2, v.z * w);
    atomicAdd(o + 3, v.w * w);
}

// ---------------------------------------------------------------------------
// BatchNorm: per-column sum / sumsq via LDS atomics, then one global atomic.
// stats[0:64)=sum, [64:128)=sumsq, [128:192)=scale, [192:256)=shift
// ---------------------------------------------------------------------------
__global__ void bnstats_k(const float* __restrict__ agg1, float* __restrict__ stats) {
    __shared__ float ssum[64], ssq[64];
    int t = threadIdx.x;
    if (t < 64) { ssum[t] = 0.0f; ssq[t] = 0.0f; }
    __syncthreads();
    for (int idx = blockIdx.x * blockDim.x + t; idx < N_NODES * HID_DIM;
         idx += gridDim.x * blockDim.x) {
        float v = agg1[idx];
        atomicAdd(&ssum[idx & 63], v);
        atomicAdd(&ssq[idx & 63], v * v);
    }
    __syncthreads();
    if (t < 64) {
        atomicAdd(&stats[t], ssum[t]);
        atomicAdd(&stats[64 + t], ssq[t]);
    }
}

__global__ void bnfinal_k(float* __restrict__ stats, const float* __restrict__ gamma,
                          const float* __restrict__ beta) {
    int t = threadIdx.x;
    if (t >= 64) return;
    float inv_n = 1.0f / (float)N_NODES;
    float mean = stats[t] * inv_n;
    float var  = stats[64 + t] * inv_n - mean * mean;
    float sc   = gamma[t] * rsqrtf(var + BN_EPS);
    stats[128 + t] = sc;
    stats[192 + t] = beta[t] - mean * sc;
}

// ---------------------------------------------------------------------------
// Fused BN + ReLU + (h @ W2) + self-loop init of output (+ b2).
// ---------------------------------------------------------------------------
__global__ void gemm2_k(const float* __restrict__ agg1, const float* __restrict__ stats,
                        const float* __restrict__ dinv, const float* __restrict__ W2,
                        const float* __restrict__ b2, float* __restrict__ t2,
                        float* __restrict__ out) {
    int i = blockIdx.x * blockDim.x + threadIdx.x;
    if (i >= N_NODES) return;
    const float4* row = (const float4*)(agg1 + (size_t)i * HID_DIM);
    float o0 = 0.0f, o1 = 0.0f;
    #pragma unroll
    for (int q = 0; q < 16; ++q) {
        float4 v = row[q];
        #pragma unroll
        for (int k = 0; k < 4; ++k) {
            int d = q * 4 + k;
            float hv = (k == 0 ? v.x : k == 1 ? v.y : k == 2 ? v.z : v.w);
            float y = hv * stats[128 + d] + stats[192 + d];
            y = fmaxf(y, 0.0f);
            o0 += y * W2[d * 2 + 0];
            o1 += y * W2[d * 2 + 1];
        }
    }
    t2[i * 2 + 0] = o0;
    t2[i * 2 + 1] = o1;
    float w = dinv[i] * dinv[i];
    out[i * 2 + 0] = o0 * w + b2[0];
    out[i * 2 + 1] = o1 * w + b2[1];
}

__global__ void scatter2_k(const int* __restrict__ src, const int* __restrict__ dst,
                           const float* __restrict__ dinv,
                           const float* __restrict__ t2, float* __restrict__ out) {
    int e = blockIdx.x * blockDim.x + threadIdx.x;
    if (e >= N_EDGES) return;
    int s = src[e], d = dst[e];
    float w = dinv[s] * dinv[d];
    atomicAdd(&out[d * 2 + 0], t2[s * 2 + 0] * w);
    atomicAdd(&out[d * 2 + 1], t2[s * 2 + 1] * w);
}

// ---------------------------------------------------------------------------
extern "C" void kernel_launch(void* const* d_in, const int* in_sizes, int n_in,
                              void* d_out, int out_size, void* d_ws, size_t ws_size,
                              hipStream_t stream) {
    const float* x     = (const float*)d_in[0];
    const int*   ei    = (const int*)d_in[1];
    const float* W1    = (const float*)d_in[2];
    const float* b1    = (const float*)d_in[3];
    const float* gamma = (const float*)d_in[4];
    const float* beta  = (const float*)d_in[5];
    const float* W2    = (const float*)d_in[6];
    const float* b2    = (const float*)d_in[7];
    const int* src = ei;
    const int* dst = ei + N_EDGES;
    float* out = (float*)d_out;

    // workspace carve-out (floats): ~26.4 MB total
    float* ws    = (float*)d_ws;
    float* deg   = ws;               // 50000
    float* dinv  = ws + 50000;       // 50000
    float* h1    = ws + 100000;      // 3,200,000
    float* agg1  = ws + 3300000;     // 3,200,000
    float* t2    = ws + 6500000;     // 100,000
    float* stats = ws + 6600000;     // 256

    zero_k    <<<(N_NODES + 255) / 256, 256, 0, stream>>>(deg, stats);
    gemm1_wmma_k<<<(TILES_M + 7) / 8, 256, 0, stream>>>(x, W1, h1);
    deg_k     <<<(N_EDGES + 255) / 256, 256, 0, stream>>>(dst, deg);
    dinv_k    <<<(N_NODES + 255) / 256, 256, 0, stream>>>(deg, dinv);
    selfinit_k<<<(N_NODES * HID_DIM + 255) / 256, 256, 0, stream>>>(h1, dinv, b1, agg1);
    scatter1_k<<<(N_EDGES * 16 + 255) / 256, 256, 0, stream>>>(src, dst, dinv, h1, agg1);
    bnstats_k <<<512, 256, 0, stream>>>(agg1, stats);
    bnfinal_k <<<1, 64, 0, stream>>>(stats, gamma, beta);
    gemm2_k   <<<(N_NODES + 255) / 256, 256, 0, stream>>>(agg1, stats, dinv, W2, b2, t2, out);
    scatter2_k<<<(N_EDGES + 255) / 256, 256, 0, stream>>>(src, dst, dinv, t2, out);
}
